// Attention_13503377178668
// MI455X (gfx1250) — matrix operand order repaired and checked by reading, not compiled, not run on hardware
//
#include <hip/hip_runtime.h>
#include <cstdint>

typedef __bf16 bf16;
typedef __attribute__((ext_vector_type(16))) __bf16 v16bf;
typedef __attribute__((ext_vector_type(8)))  __bf16 v8bf;
typedef __attribute__((ext_vector_type(8)))  float  v8f;
typedef __attribute__((ext_vector_type(4)))  unsigned u32x4;
typedef __attribute__((ext_vector_type(8)))  int      i32x8;
typedef __attribute__((ext_vector_type(4)))  int      i32x4;

__device__ __forceinline__ v8f wmma_bf16(v16bf a, v16bf b, v8f c) {
  // D = A(16x32 bf16) * B(32x16 bf16) + C(16x16 f32)
  return __builtin_amdgcn_wmma_f32_16x16x32_bf16(
      /*neg_a=*/false, a, /*neg_b=*/false, b,
      /*c_mod=*/(short)0, c, /*reuse_a=*/false, /*reuse_b=*/false);
}

// A fragment: 16x32 bf16. rowptr -> element [m][0] for m = lane&15.
// lanes 0-15: kbase=0 ; lanes 16-31: kbase=8.
// elem e<8 -> k=kbase+e ; e>=8 -> k=kbase+16+(e-8). Two 16B loads.
__device__ __forceinline__ v16bf load_a_frag(const bf16* rowptr, int lane) {
  const int kb = (lane < 16) ? 0 : 8;
  v8bf lo = *(const v8bf*)(rowptr + kb);
  v8bf hi = *(const v8bf*)(rowptr + kb + 16);
  v16bf r;
#pragma unroll
  for (int e = 0; e < 8; ++e) { r[e] = lo[e]; r[e + 8] = hi[e]; }
  return r;
}

// B fragment: 32x16 bf16 stored n-major ([n][k] contiguous in k).
// colptr -> element [n][0] for n = lane&15.
// lanes 0-15: kbase=0 ; lanes 16-31: kbase=8. e<8 -> k=kbase+e ; e>=8 -> k=kbase+16+(e-8). Two 16B loads.
__device__ __forceinline__ v16bf load_b_frag(const bf16* colptr, int lane) {
  const int kb = (lane < 16) ? 0 : 8;
  v8bf lo = *(const v8bf*)(colptr + kb);
  v8bf hi = *(const v8bf*)(colptr + kb + 16);
  v16bf r;
#pragma unroll
  for (int e = 0; e < 8; ++e) { r[e] = lo[e]; r[e + 8] = hi[e]; }
  return r;
}

// ---------------------------------------------------------------------------
// Tensor Data Mover: 2D tile load Global -> LDS (bf16 data, data_size=2B)
// D# per CDNA5 ISA. 6-arg builtin (clang-23 form).
// pad_interval_code: 2^(c+1) DWORDs between pads; pad_amount_code: c+1 DWORDs.
// ---------------------------------------------------------------------------
#if defined(__has_builtin) && __has_builtin(__builtin_amdgcn_tensor_load_to_lds)
#define HAVE_TDM 1
#else
#define HAVE_TDM 0
#endif

__device__ __forceinline__ void tdm_load_2d_bf16(
    unsigned lds_addr, const void* gtile,
    unsigned tensor_d0, unsigned tensor_d1,
    unsigned tile_d0, unsigned tile_d1,
    unsigned stride_d0,
    unsigned pad_interval_code, unsigned pad_amount_code)
{
#if HAVE_TDM
  unsigned long long ga = (unsigned long long)(uintptr_t)gtile;
  u32x4 g0;
  g0[0] = 1u;                                            // count=1, user mode
  g0[1] = lds_addr;                                      // LDS byte address
  g0[2] = (unsigned)(ga & 0xffffffffu);                  // global_addr[31:0]
  g0[3] = (unsigned)((ga >> 32) & 0x1ffffffu) | (2u << 30); // [56:32] | type=2
  i32x8 g1;
  unsigned w0 = (1u << 16)                               // data_size = 2 bytes
              | (1u << 20)                               // pad_enable
              | (pad_interval_code << 22)
              | (pad_amount_code << 25);
  g1[0] = (int)w0;
  g1[1] = (int)((tensor_d0 & 0xffffu) << 16);            // tensor_dim0[15:0]
  g1[2] = (int)(((tensor_d0 >> 16) & 0xffffu) | ((tensor_d1 & 0xffffu) << 16));
  g1[3] = (int)(((tensor_d1 >> 16) & 0xffffu) | ((tile_d0 & 0xffffu) << 16));
  g1[4] = (int)(tile_d1 & 0xffffu);                      // tile_dim1; tile_dim2=0
  g1[5] = (int)stride_d0;                                // dim0 stride [31:0]
  g1[6] = 0;                                             // stride hi, dim1 stride lo
  g1[7] = 0;
  i32x4 z4 = {0, 0, 0, 0};
  i32x8 z8 = {0, 0, 0, 0, 0, 0, 0, 0};
  __builtin_amdgcn_tensor_load_to_lds(g0, g1, z4, z4, z8, 0);
#else
  (void)lds_addr; (void)gtile; (void)tensor_d0; (void)tensor_d1;
  (void)tile_d0; (void)tile_d1; (void)stride_d0;
  (void)pad_interval_code; (void)pad_amount_code;
#endif
}

__device__ __forceinline__ unsigned lds_offset_of(const void* p) {
  return (unsigned)(uintptr_t)p;   // generic LDS ptr: addr[31:0] = LDS offset
}

// ---------------------------------------------------------------------------
// Projection GEMM: C[M,N] = A[M,K](f32) @ W[K,N](f32), output bf16 in
// head-major layout (b, h, row_in_batch, d). If outV != nullptr, columns
// >= N/2 go to outV (kv -> k/v split). rowShift: log2(rows per batch).
// Block: 128 threads (4 waves). Tile: 64(M) x 64(N), K-step 32.
// ---------------------------------------------------------------------------
__global__ __launch_bounds__(128) void proj_kernel(
    const float* __restrict__ A, const float* __restrict__ W,
    bf16* __restrict__ outK, bf16* __restrict__ outV,
    int M, int K, int N, int rowShift)
{
  __shared__ bf16 As[64][40];   // [m][k] padded: 80B rows, 16B-aligned frags
  __shared__ bf16 Bs[64][40];   // [n][k] padded

  const int tid  = threadIdx.x;
  const int lane = tid & 31;
  const int wave = tid >> 5;
  const int nTilesN = N >> 6;
  const int m0 = (blockIdx.x / nTilesN) * 64;
  const int n0 = (blockIdx.x % nTilesN) * 64;

  v8f acc[4] = {};

  for (int k0 = 0; k0 < K; k0 += 32) {
    // A tile 64x32 f32 -> bf16 LDS (batched float4 global reads)
    float4 va[4];
#pragma unroll
    for (int it = 0; it < 4; ++it) {
      int idx = tid + it * 128;
      int r = idx >> 3, c4 = (idx & 7) * 4;
      va[it] = *(const float4*)&A[(size_t)(m0 + r) * K + k0 + c4];
    }
    // W tile 32x64 f32 (batched float4 reads), stored transposed [n][k]
    float4 vb[4];
#pragma unroll
    for (int it = 0; it < 4; ++it) {
      int idx = tid + it * 128;
      int k = idx >> 4, n4 = (idx & 15) * 4;
      vb[it] = *(const float4*)&W[(size_t)(k0 + k) * N + n0 + n4];
    }
#pragma unroll
    for (int it = 0; it < 4; ++it) {
      int idx = tid + it * 128;
      int r = idx >> 3, c4 = (idx & 7) * 4;
      As[r][c4 + 0] = (bf16)va[it].x; As[r][c4 + 1] = (bf16)va[it].y;
      As[r][c4 + 2] = (bf16)va[it].z; As[r][c4 + 3] = (bf16)va[it].w;
    }
#pragma unroll
    for (int it = 0; it < 4; ++it) {
      int idx = tid + it * 128;
      int k = idx >> 4, n4 = (idx & 15) * 4;
      Bs[n4 + 0][k] = (bf16)vb[it].x; Bs[n4 + 1][k] = (bf16)vb[it].y;
      Bs[n4 + 2][k] = (bf16)vb[it].z; Bs[n4 + 3][k] = (bf16)vb[it].w;
    }
    __syncthreads();
    v16bf a = load_a_frag(&As[wave * 16 + (lane & 15)][0], lane);
#pragma unroll
    for (int nt = 0; nt < 4; ++nt) {
      v16bf b = load_b_frag(&Bs[nt * 16 + (lane & 15)][0], lane);
      acc[nt] = wmma_bf16(a, b, acc[nt]);
    }
    __syncthreads();
  }

  // Epilogue: f32 acc -> bf16, head-major scatter.
  const int halfN = N >> 1;
  const int rowsMask = (1 << rowShift) - 1;
#pragma unroll
  for (int nt = 0; nt < 4; ++nt) {
#pragma unroll
    for (int r = 0; r < 8; ++r) {
      int gm = m0 + wave * 16 + r + ((lane >> 4) << 3);
      int gc = n0 + nt * 16 + (lane & 15);
      bf16* dst = outK;
      int c = gc;
      if (outV != nullptr && c >= halfN) { c -= halfN; dst = outV; }
      int h = c >> 6, d = c & 63;
      int b = gm >> rowShift, i = gm & rowsMask;
      dst[(((size_t)(b * 16 + h) << rowShift) + i) * 64 + d] = (bf16)acc[nt][r];
    }
  }
}

// ---------------------------------------------------------------------------
// Attention: one block per (b, h, 16-row query chunk). 128 threads / 4 waves.
// Phase 1: S = scale * Q Kt  (WMMA, K chunks of 128 j TDM-loaded into LDS)
// Phase 2: softmax over full J=512 in LDS
// Phase 3: O = P V          (WMMA, V chunks staged transposed in LDS)
// ---------------------------------------------------------------------------
__global__ __launch_bounds__(128) void attn_kernel(
    const bf16* __restrict__ qh, const bf16* __restrict__ kh,
    const bf16* __restrict__ vh, bf16* __restrict__ attnOut)
{
  __shared__ float sc[16][516];        // scores, padded (stride 516 f32)
  __shared__ bf16  kvbuf[128 * 80];    // K chunk [j][80] / V-transposed [d][136]
  __shared__ float red[16][8];
  __shared__ float rowmax[16];
  __shared__ float rowsum[16];

  const int tid  = threadIdx.x;
  const int lane = tid & 31;
  const int wave = tid >> 5;
  const int ic = blockIdx.x & 7;        // query chunk (16 rows each)
  const int bh = blockIdx.x >> 3;       // b*16 + h
  const size_t qbase  = ((size_t)bh * 128 + ic * 16) * 64;
  const size_t kvbase = (size_t)bh * 512 * 64;
  const float scale = 0.125f;           // 64^-0.5

  // Q fragments held in registers (loaded straight from global, bf16)
  v16bf qf[2];
  {
    const bf16* qrow = qh + qbase + (size_t)(lane & 15) * 64;
    qf[0] = load_a_frag(qrow, lane);
    qf[1] = load_a_frag(qrow + 32, lane);
  }

  // ---- Phase 1: scores ----
  for (int jc = 0; jc < 4; ++jc) {
#if HAVE_TDM
    if (wave == 0) {
      // TDM: 128x64 bf16 tile, rows padded 64 -> 80 elems (128B + 32B pad):
      // pad_interval code 4 (32 DWORDs), pad_amount code 7 (8 DWORDs).
      tdm_load_2d_bf16(lds_offset_of(&kvbuf[0]),
                       kh + kvbase + (size_t)jc * 128 * 64,
                       /*tensor_d0=*/64, /*tensor_d1=*/128,
                       /*tile_d0=*/64, /*tile_d1=*/128,
                       /*stride_d0=*/64,
                       /*pad_interval=*/4, /*pad_amount=*/7);
      __builtin_amdgcn_s_wait_tensorcnt(0);
    }
#else
    {
      const bf16* src = kh + kvbase + ((size_t)jc * 128 + tid) * 64;
#pragma unroll
      for (int q8 = 0; q8 < 8; ++q8)
        *(v8bf*)&kvbuf[tid * 80 + q8 * 8] = *(const v8bf*)(src + q8 * 8);
    }
#endif
    __syncthreads();
#pragma unroll
    for (int jt2 = 0; jt2 < 2; ++jt2) {
      int jt = wave * 2 + jt2;
      v8f s = {};
#pragma unroll
      for (int dch = 0; dch < 2; ++dch) {
        v16bf b = load_b_frag(&kvbuf[(jt * 16 + (lane & 15)) * 80 + dch * 32], lane);
        s = wmma_bf16(qf[dch], b, s);
      }
#pragma unroll
      for (int r = 0; r < 8; ++r) {
        int m = r + ((lane >> 4) << 3);
        sc[m][jc * 128 + jt * 16 + (lane & 15)] = s[r] * scale;
      }
    }
    __syncthreads();
  }

  // ---- Phase 2: softmax over J=512 ----
  {
    int row = tid >> 3, seg = tid & 7;
    float* sr = &sc[row][seg * 64];
    float mx = -1e30f;
    for (int c = 0; c < 64; ++c) mx = fmaxf(mx, sr[c]);
    red[row][seg] = mx;
    __syncthreads();
    if (seg == 0) {
      float m2 = red[row][0];
      for (int s2 = 1; s2 < 8; ++s2) m2 = fmaxf(m2, red[row][s2]);
      rowmax[row] = m2;
    }
    __syncthreads();
    float rm = rowmax[row];
    float sum = 0.f;
    for (int c = 0; c < 64; ++c) { float e = __expf(sr[c] - rm); sr[c] = e; sum += e; }
    red[row][seg] = sum;
    __syncthreads();
    if (seg == 0) {
      float t = 0.f;
      for (int s2 = 0; s2 < 8; ++s2) t += red[row][s2];
      rowsum[row] = t;
    }
    __syncthreads();
    float inv = 1.0f / rowsum[row];
    for (int c = 0; c < 64; ++c) sr[c] *= inv;
  }
  __syncthreads();

  // ---- Phase 3: O = P V  (wave owns d-tile = wave*16..+15) ----
  v8f o = {};
  for (int jc = 0; jc < 4; ++jc) {
    {
      // stage V chunk transposed: VT[d][j] at kvbuf[d*136 + j]
      const bf16* src = vh + kvbase + ((size_t)jc * 128 + tid) * 64;
      v8bf vv[8];
#pragma unroll
      for (int q8 = 0; q8 < 8; ++q8) vv[q8] = *(const v8bf*)(src + q8 * 8);
#pragma unroll
      for (int q8 = 0; q8 < 8; ++q8) {
#pragma unroll
        for (int e = 0; e < 8; ++e) kvbuf[(q8 * 8 + e) * 136 + tid] = vv[q8][e];
      }
    }
    __syncthreads();
#pragma unroll
    for (int g = 0; g < 4; ++g) {
      v16bf p;
      {
        int m = lane & 15;
        int kb = (lane < 16) ? 0 : 8;
        const float* srow = &sc[m][jc * 128 + g * 32 + kb];
#pragma unroll
        for (int e = 0; e < 8; ++e) {
          p[e]     = (bf16)srow[e];
          p[e + 8] = (bf16)srow[e + 16];
        }
      }
      v16bf b = load_b_frag(&kvbuf[(wave * 16 + (lane & 15)) * 136 + g * 32], lane);
      o = wmma_bf16(p, b, o);
    }
    __syncthreads();
  }

  // Epilogue: attnOut row-major (B*I, 1024) bf16
  {
    int b = bh >> 4, h = bh & 15;
#pragma unroll
    for (int r = 0; r < 8; ++r) {
      int i = ic * 16 + r + ((lane >> 4) << 3);
      int d = wave * 16 + (lane & 15);
      attnOut[((size_t)(b * 128 + i)) * 1024 + h * 64 + d] = (bf16)o[r];
    }
  }
}

// ---------------------------------------------------------------------------
// Output projection: out[M,N](f32) = A[M,K](bf16) @ W[K,N](f32) + bias[N]
// A tile staged via TDM (pure bf16 copy); W tile converted f32->bf16 by VALU.
// ---------------------------------------------------------------------------
__global__ __launch_bounds__(128) void oproj_kernel(
    const bf16* __restrict__ A, const float* __restrict__ W,
    const float* __restrict__ bias, float* __restrict__ out,
    int M, int K, int N)
{
  __shared__ bf16 As[64][40];
  __shared__ bf16 Bs[64][40];

  const int tid  = threadIdx.x;
  const int lane = tid & 31;
  const int wave = tid >> 5;
  const int nTilesN = N >> 6;
  const int m0 = (blockIdx.x / nTilesN) * 64;
  const int n0 = (blockIdx.x % nTilesN) * 64;

  v8f acc[4] = {};

  for (int k0 = 0; k0 < K; k0 += 32) {
#if HAVE_TDM
    if (wave == 0) {
      // TDM: 64x32 bf16 tile, rows padded 32 -> 40 elems (64B + 16B pad):
      // pad_interval code 3 (16 DWORDs), pad_amount code 3 (4 DWORDs).
      tdm_load_2d_bf16(lds_offset_of(&As[0][0]),
                       A + (size_t)m0 * K + k0,
                       /*tensor_d0=*/32, /*tensor_d1=*/64,
                       /*tile_d0=*/32, /*tile_d1=*/64,
                       /*stride_d0=*/(unsigned)K,
                       /*pad_interval=*/3, /*pad_amount=*/3);
    }
#else
#pragma unroll
    for (int it = 0; it < 2; ++it) {
      int idx = tid + it * 128;
      int r = idx >> 2, c8 = (idx & 3) * 8;
      *(v8bf*)&As[r][c8] = *(const v8bf*)&A[(size_t)(m0 + r) * K + k0 + c8];
    }
#endif
    // W tile 32x64 f32 (batched reads), stored transposed [n][k]
    float4 vb[4];
#pragma unroll
    for (int it = 0; it < 4; ++it) {
      int idx = tid + it * 128;
      int k = idx >> 4, n4 = (idx & 15) * 4;
      vb[it] = *(const float4*)&W[(size_t)(k0 + k) * N + n0 + n4];
    }
#pragma unroll
    for (int it = 0; it < 4; ++it) {
      int idx = tid + it * 128;
      int k = idx >> 4, n4 = (idx & 15) * 4;
      Bs[n4 + 0][k] = (bf16)vb[it].x; Bs[n4 + 1][k] = (bf16)vb[it].y;
      Bs[n4 + 2][k] = (bf16)vb[it].z; Bs[n4 + 3][k] = (bf16)vb[it].w;
    }
#if HAVE_TDM
    if (wave == 0) __builtin_amdgcn_s_wait_tensorcnt(0);
#endif
    __syncthreads();
    v16bf a = load_a_frag(&As[wave * 16 + (lane & 15)][0], lane);
#pragma unroll
    for (int nt = 0; nt < 4; ++nt) {
      v16bf b = load_b_frag(&Bs[nt * 16 + (lane & 15)][0], lane);
      acc[nt] = wmma_bf16(a, b, acc[nt]);
    }
    __syncthreads();
  }

#pragma unroll
  for (int nt = 0; nt < 4; ++nt) {
#pragma unroll
    for (int r = 0; r < 8; ++r) {
      int gm = m0 + wave * 16 + r + ((lane >> 4) << 3);
      int gn = n0 + nt * 16 + (lane & 15);
      out[(size_t)gm * N + gn] = acc[nt][r] + bias[gn];
    }
  }
}

// ---------------------------------------------------------------------------
extern "C" void kernel_launch(void* const* d_in, const int* in_sizes, int n_in,
                              void* d_out, int out_size, void* d_ws, size_t ws_size,
                              hipStream_t stream) {
  (void)in_sizes; (void)n_in; (void)out_size; (void)ws_size;
  const float* q   = (const float*)d_in[0];  // (64,128,512)
  const float* kv  = (const float*)d_in[1];  // (64,512,256)
  const float* Wq  = (const float*)d_in[2];  // (512,1024)
  const float* Wkv = (const float*)d_in[3];  // (256,2048)
  const float* Wo  = (const float*)d_in[4];  // (1024,512)
  const float* bo  = (const float*)d_in[5];  // (512)
  float* out = (float*)d_out;                // (64,128,512)

  // Workspace layout (bf16): qh | kh | vh | attnOut
  bf16* qh   = (bf16*)d_ws;
  bf16* kh   = qh + (size_t)64 * 16 * 128 * 64;
  bf16* vh   = kh + (size_t)64 * 16 * 512 * 64;
  bf16* aout = vh + (size_t)64 * 16 * 512 * 64;

  // 1) query = q @ Wq   (M=8192, K=512, N=1024), rows/batch = 128 (shift 7)
  proj_kernel<<<dim3((8192 / 64) * (1024 / 64)), 128, 0, stream>>>(
      q, Wq, qh, nullptr, 8192, 512, 1024, 7);

  // 2) k,v = kv @ Wkv   (M=32768, K=256, N=2048), rows/batch = 512 (shift 9)
  proj_kernel<<<dim3((32768 / 64) * (2048 / 64)), 128, 0, stream>>>(
      kv, Wkv, kh, vh, 32768, 256, 2048, 9);

  // 3) attention: one block per (b,h,ic) = 64*16*8
  attn_kernel<<<dim3(64 * 16 * 8), 128, 0, stream>>>(qh, kh, vh, aout);

  // 4) out = attnOut @ Wo + bo  (M=8192, K=1024, N=512)
  oproj_kernel<<<dim3((8192 / 64) * (512 / 64)), 128, 0, stream>>>(
      aout, Wo, bo, out, 8192, 1024, 512);
}
